// OctreeAttention_87935160418650
// MI455X (gfx1250) — matrix operand order repaired and checked by reading, not compiled, not run on hardware
//
#include <hip/hip_runtime.h>

// ---------------- problem constants ----------------
#define N_TOK    400000
#define C_DIM    256
#define BLK_TOK  128
#define NBLKS    (N_TOK / BLK_TOK)      // 3125 (exact)
#define SCALE_F  0.17677669529663687f   // 32^-0.5
#define INVALID_F (-1000.0f)

typedef __attribute__((ext_vector_type(16))) _Float16 v16h;
typedef __attribute__((ext_vector_type(8)))  _Float16 v8h;
typedef __attribute__((ext_vector_type(8)))  float    v8f;
typedef unsigned int v4u __attribute__((ext_vector_type(4)));
typedef int          v8i __attribute__((ext_vector_type(8)));
typedef int          v4i __attribute__((ext_vector_type(4)));

// ---------------- LDS layout (bytes), all 16B aligned ----------------
#define QSTR   776                       // qkv row stride in f16 (768+8)
#define SOSTR  264                       // data/out row stride in f16 (256+8)
#define PSTR   40                        // probs row stride in f16 (32+8)
#define QKV_OFF 0
#define OUT_OFF (BLK_TOK*QSTR*2)                 // 198656
#define P_OFF   (OUT_OFF + BLK_TOK*SOSTR*2)      // 266240
#define XYZ_OFF (P_OFF + 8*32*PSTR*2)            // 286720
#define FRQ_OFF (XYZ_OFF + BLK_TOK*4*4)          // 288768
#define DEP_OFF (FRQ_OFF + 8*16*3*4)             // 290304
#define LDS_BYTES (DEP_OFF + BLK_TOK*4)          // 290816  (< 320KB WGP LDS)

// ---------------- helpers ----------------
__device__ __forceinline__ v8f wmma16(v16h a, v16h b, v8f c) {
  // D = A(16x32 f16) * B(32x16 f16) + C(16x16 f32)
  return __builtin_amdgcn_wmma_f32_16x16x32_f16(false, a, false, b, (short)0, c,
                                                false, false);
}

__device__ __forceinline__ v16h mkfrag(v8h lo, v8h hi) {
  return __builtin_shufflevector(lo, hi, 0, 1, 2, 3, 4, 5, 6, 7, 8, 9, 10, 11,
                                 12, 13, 14, 15);
}

__device__ __forceinline__ float rmax16(float x) {
  #pragma unroll
  for (int o = 1; o < 16; o <<= 1) x = fmaxf(x, __shfl_xor(x, o, 32));
  return x;
}
__device__ __forceinline__ float rsum16(float x) {
  #pragma unroll
  for (int o = 1; o < 16; o <<= 1) x += __shfl_xor(x, o, 32);
  return x;
}

// Accumulate one 16x16 output tile over K=256 (8 WMMA steps).
// Wt is f16 column-major [cols][256]; col already includes the lane's column.
__device__ __forceinline__ v8f gemm_col_tile(const v16h afr[8],
                                             const _Float16* __restrict__ Wt,
                                             int col, int lhalf) {
  v8f acc = {0.f, 0.f, 0.f, 0.f, 0.f, 0.f, 0.f, 0.f};
  const _Float16* base = Wt + (size_t)col * 256 + lhalf * 8;
  #pragma unroll
  for (int kc = 0; kc < 8; ++kc) {
    v16h b = mkfrag(*(const v8h*)(base + kc * 32), *(const v8h*)(base + kc * 32 + 16));
    acc = wmma16(afr[kc], b, acc);
  }
  return acc;
}

// ---------------- prep: convert + transpose weights to f16 ----------------
__global__ void prep_weights(const float* __restrict__ Wqkv,
                             const float* __restrict__ Wproj,
                             _Float16* __restrict__ WqkvT,
                             _Float16* __restrict__ WprojT) {
  int i = blockIdx.x * blockDim.x + threadIdx.x;
  if (i < 256 * 768) {                 // Wqkv [256][768] -> WqkvT [768][256]
    int k = i / 768, c = i % 768;
    WqkvT[(size_t)c * 256 + k] = (_Float16)Wqkv[i];
  }
  if (i < 256 * 256) {                 // Wproj [256][256] -> WprojT [256][256]
    int k = i / 256, c = i % 256;
    WprojT[(size_t)c * 256 + k] = (_Float16)Wproj[i];
  }
}

// ---------------- fused per-block kernel ----------------
__global__ __launch_bounds__(256, 1)
void octree_attn_fused(const float* __restrict__ data,
                       const float* __restrict__ xyz,
                       const int* __restrict__ depth,
                       const float* __restrict__ bqkv,
                       const float* __restrict__ bproj,
                       const float* __restrict__ freqs,
                       const _Float16* __restrict__ WqkvT,
                       const _Float16* __restrict__ WprojT,
                       float* __restrict__ out) {
  extern __shared__ char smem[];
  _Float16* sQKV  = (_Float16*)(smem + QKV_OFF);   // [128][QSTR]
  _Float16* sData = (_Float16*)(smem + OUT_OFF);   // [128][SOSTR] (reused as sOut)
  _Float16* sOut  = sData;
  _Float16* sP    = (_Float16*)(smem + P_OFF);     // [8 waves][32][PSTR]
  float*    sXYZ  = (float*)(smem + XYZ_OFF);      // [128][4]
  float*    sFrq  = (float*)(smem + FRQ_OFF);      // [8][16][3]
  int*      sDep  = (int*)(smem + DEP_OFF);        // [128]

  const int blk   = blockIdx.x;
  const int tid   = threadIdx.x;
  const int wave  = tid >> 5;
  const int lane  = tid & 31;
  const int lrow  = lane & 15;   // column-in-tile for B/C/D, row for A
  const int lhalf = lane >> 4;   // which 16-lane half
  const int m0    = wave * 16;   // this wave's 16-row strip

  // ---------- Phase 1: TDM the f32 data tile into LDS, stage sides ----------
  // The 128KB f32 tile lands in the (currently dead) sQKV region via the
  // Tensor Data Mover; waves meanwhile fetch xyz/depth/freqs. After the DMA
  // completes, all threads convert f32 -> f16 into the padded sData buffer.
  if (wave == 0) {
    unsigned lds_base = (unsigned)(size_t)(void*)smem;   // sQKV at offset 0
    unsigned long long ga =
        (unsigned long long)(size_t)(const void*)(data +
                                                  (size_t)blk * BLK_TOK * C_DIM);
    v4u g0;
    g0[0] = 1u;                                   // count=1, user descriptor
    g0[1] = lds_base;                             // lds_addr
    g0[2] = (unsigned)ga;                         // global_addr[31:0]
    g0[3] = (unsigned)(ga >> 32) | (2u << 30);    // global_addr[56:32] | type=2
    v8i g1;
    g1[0] = (2 << 16);                            // data_size=4B; no pad/iter
    g1[1] = (int)(256u << 16);                    // tensor_dim0 = 256 (bits 63:48)
    g1[2] = (int)(128u << 16);                    // tensor_dim1 = 128 (bits 95:80)
    g1[3] = (int)(256u << 16);                    // tile_dim0 = 256 (bits 127:112)
    g1[4] = 128;                                  // tile_dim1 = 128
    g1[5] = 256;                                  // tensor_dim0_stride = 256
    g1[6] = 0;
    g1[7] = 0;
    v4i g2 = {0, 0, 0, 0};
    v4i g3 = {0, 0, 0, 0};
    v8i g4 = {0, 0, 0, 0, 0, 0, 0, 0};            // unused group (2-D tile)
    __builtin_amdgcn_tensor_load_to_lds(g0, g1, g2, g3, g4, 0);
  }
  if (tid < BLK_TOK) {
    const float* xp = xyz + (size_t)(blk * BLK_TOK + tid) * 3;
    sXYZ[tid * 4 + 0] = xp[0];
    sXYZ[tid * 4 + 1] = xp[1];
    sXYZ[tid * 4 + 2] = xp[2];
    sDep[tid] = depth[blk * BLK_TOK + tid];
  }
  for (int idx = tid; idx < 8 * 16 * 3; idx += 256) sFrq[idx] = freqs[idx];
  if (wave == 0) __builtin_amdgcn_s_wait_tensorcnt(0);
  __syncthreads();
  {
    const float* stage = (const float*)(smem + QKV_OFF);
    for (int idx = tid; idx < BLK_TOK * C_DIM; idx += 256) {
      int r = idx >> 8, c = idx & 255;
      sData[r * SOSTR + c] = (_Float16)stage[idx];
    }
  }
  __syncthreads();

  // ---------- Phase 2: QKV GEMM (WMMA) + bias + fused RoPE ----------
  {
    // Load all 8 A-fragments for this wave's 16-row strip (resident in VGPRs).
    v16h afr[8];
    const _Float16* arow = sData + (size_t)(m0 + lrow) * SOSTR;
    #pragma unroll
    for (int kc = 0; kc < 8; ++kc) {
      const _Float16* p = arow + kc * 32 + lhalf * 8;
      afr[kc] = mkfrag(*(const v8h*)p, *(const v8h*)(p + 16));
    }

    // Q (cols 0..255) and K (cols 256..511): process head halves in pairs
    // so RoPE (rotate-half) can be applied in registers.
    #pragma unroll 1
    for (int reg = 0; reg < 2; ++reg) {
      int colbase = reg * 256;
      #pragma unroll 1
      for (int h = 0; h < 8; ++h) {
        int c0 = colbase + h * 32;
        __builtin_prefetch(WqkvT + (size_t)(c0 + 32) * 256, 0, 1);
        v8f a0 = gemm_col_tile(afr, WqkvT, c0 + lrow, lhalf);
        v8f a1 = gemm_col_tile(afr, WqkvT, c0 + 16 + lrow, lhalf);
        float b0 = bqkv[c0 + lrow];
        float b1 = bqkv[c0 + 16 + lrow];
        const float* fr = sFrq + (h * 16 + lrow) * 3;
        float f0 = fr[0], f1 = fr[1], f2 = fr[2];
        #pragma unroll
        for (int r = 0; r < 8; ++r) {
          int m = m0 + r + lhalf * 8;                 // token row (C-layout)
          const float* xp = sXYZ + m * 4;
          float th = xp[0] * f0 + xp[1] * f1 + xp[2] * f2;
          float sn, cs;
          __sincosf(th, &sn, &cs);
          float x0 = a0[r] + b0, x1 = a1[r] + b1;
          sQKV[(size_t)m * QSTR + c0 + lrow]      = (_Float16)(x0 * cs - x1 * sn);
          sQKV[(size_t)m * QSTR + c0 + 16 + lrow] = (_Float16)(x1 * cs + x0 * sn);
        }
      }
    }
    // V (cols 512..767): straight through.
    #pragma unroll 1
    for (int nt = 32; nt < 48; ++nt) {
      int col = nt * 16 + lrow;
      __builtin_prefetch(WqkvT + (size_t)(col + 16) * 256, 0, 1);
      v8f a = gemm_col_tile(afr, WqkvT, col, lhalf);
      float b = bqkv[col];
      #pragma unroll
      for (int r = 0; r < 8; ++r) {
        int m = m0 + r + lhalf * 8;
        sQKV[(size_t)m * QSTR + col] = (_Float16)(a[r] + b);
      }
    }
  }
  __syncthreads();

  // ---------- Phase 3: windowed attention, 4 (d,h) pairs per wave ----------
  #pragma unroll 1
  for (int it = 0; it < 4; ++it) {
    int idx = wave * 4 + it;
    int d = idx >> 3, h = idx & 7;
    int qcol = h * 32, kcol = 256 + h * 32, vcol = 512 + h * 32;

    // Q A-frags: row = query (mt*16 + lrow), token = row*4 + d
    v16h aq[2];
    #pragma unroll
    for (int mt = 0; mt < 2; ++mt) {
      const _Float16* p =
          sQKV + (size_t)((mt * 16 + lrow) * 4 + d) * QSTR + qcol + lhalf * 8;
      aq[mt] = mkfrag(*(const v8h*)p, *(const v8h*)(p + 16));
    }
    // K^T B-frags: col = key (nt*16 + lrow), elements = head dims
    v16h bk[2];
    #pragma unroll
    for (int nt = 0; nt < 2; ++nt) {
      const _Float16* p =
          sQKV + (size_t)((nt * 16 + lrow) * 4 + d) * QSTR + kcol + lhalf * 8;
      bk[nt] = mkfrag(*(const v8h*)p, *(const v8h*)(p + 16));
    }
    v8f zero = {0.f, 0.f, 0.f, 0.f, 0.f, 0.f, 0.f, 0.f};
    v8f s[2][2];
    #pragma unroll
    for (int mt = 0; mt < 2; ++mt)
      #pragma unroll
      for (int nt = 0; nt < 2; ++nt) s[mt][nt] = wmma16(aq[mt], bk[nt], zero);

    int dk0 = sDep[lrow * 4 + d];
    int dk1 = sDep[(16 + lrow) * 4 + d];
    _Float16* sPw = sP + (size_t)wave * 32 * PSTR;

    #pragma unroll
    for (int mt = 0; mt < 2; ++mt) {
      #pragma unroll
      for (int r = 0; r < 8; ++r) {
        int qloc = mt * 16 + r + lhalf * 8;
        int dq = sDep[qloc * 4 + d];
        float v0 = s[mt][0][r] * SCALE_F + (dq < dk0 ? INVALID_F : 0.f);
        float v1 = s[mt][1][r] * SCALE_F + (dq < dk1 ? INVALID_F : 0.f);
        float mx = rmax16(fmaxf(v0, v1));
        float e0 = __expf(v0 - mx), e1 = __expf(v1 - mx);
        float inv = __builtin_amdgcn_rcpf(rsum16(e0 + e1));
        sPw[qloc * PSTR + lrow]      = (_Float16)(e0 * inv);
        sPw[qloc * PSTR + 16 + lrow] = (_Float16)(e1 * inv);
      }
    }

    // P A-frags (re-layout through wave-private LDS)
    v16h aP[2];
    #pragma unroll
    for (int mt = 0; mt < 2; ++mt) {
      const _Float16* p = sPw + (mt * 16 + lrow) * PSTR + lhalf * 16;
      aP[mt] = mkfrag(*(const v8h*)p, *(const v8h*)(p + 8));
    }
    // V B-frags: col = out dim, elements = keys (gather across token rows)
    v16h bV[2];
    #pragma unroll
    for (int nt = 0; nt < 2; ++nt) {
      union { v16h v; _Float16 e[16]; } u;
      #pragma unroll
      for (int j = 0; j < 16; ++j) {
        int k = lhalf * 16 + j;
        u.e[j] = sQKV[(size_t)(k * 4 + d) * QSTR + vcol + nt * 16 + lrow];
      }
      bV[nt] = u.v;
    }
    #pragma unroll
    for (int mt = 0; mt < 2; ++mt)
      #pragma unroll
      for (int nt = 0; nt < 2; ++nt) {
        v8f o = wmma16(aP[mt], bV[nt], zero);
        #pragma unroll
        for (int r = 0; r < 8; ++r) {
          int qloc = mt * 16 + r + lhalf * 8;
          sOut[(size_t)(qloc * 4 + d) * SOSTR + h * 32 + nt * 16 + lrow] =
              (_Float16)o[r];
        }
      }
  }
  __syncthreads();

  // ---------- Phase 4: output projection (WMMA) + bias, write f32 ----------
  {
    v16h af2[8];
    const _Float16* arow = sOut + (size_t)(m0 + lrow) * SOSTR;
    #pragma unroll
    for (int kc = 0; kc < 8; ++kc) {
      const _Float16* p = arow + kc * 32 + lhalf * 8;
      af2[kc] = mkfrag(*(const v8h*)p, *(const v8h*)(p + 16));
    }
    #pragma unroll 1
    for (int nt = 0; nt < 16; ++nt) {
      int col = nt * 16 + lrow;
      __builtin_prefetch(WprojT + (size_t)(col + 16) * 256, 0, 1);
      v8f acc = gemm_col_tile(af2, WprojT, col, lhalf);
      float b = bproj[col];
      #pragma unroll
      for (int r = 0; r < 8; ++r) {
        int m = m0 + r + lhalf * 8;
        out[((size_t)blk * BLK_TOK + m) * C_DIM + col] = acc[r] + b;
      }
    }
  }
}

// ---------------- launch ----------------
extern "C" void kernel_launch(void* const* d_in, const int* in_sizes, int n_in,
                              void* d_out, int out_size, void* d_ws,
                              size_t ws_size, hipStream_t stream) {
  (void)in_sizes; (void)n_in; (void)out_size; (void)ws_size;
  const float* data  = (const float*)d_in[0];
  const float* xyz   = (const float*)d_in[1];
  const int*   depth = (const int*)d_in[2];
  const float* Wqkv  = (const float*)d_in[3];
  const float* bqkv  = (const float*)d_in[4];
  const float* Wproj = (const float*)d_in[5];
  const float* bproj = (const float*)d_in[6];
  const float* freqs = (const float*)d_in[7];
  float* out = (float*)d_out;

  _Float16* WqkvT  = (_Float16*)d_ws;                  // [768][256] f16
  _Float16* WprojT = WqkvT + (size_t)768 * 256;        // [256][256] f16

  prep_weights<<<(256 * 768 + 255) / 256, 256, 0, stream>>>(Wqkv, Wproj, WqkvT,
                                                            WprojT);
  octree_attn_fused<<<NBLKS, 256, LDS_BYTES, stream>>>(
      data, xyz, depth, bqkv, bproj, freqs, WqkvT, WprojT, out);
}
